// MoE_56427280335535
// MI455X (gfx1250) — matrix operand order repaired and checked by reading, not compiled, not run on hardware
//
#include <hip/hip_runtime.h>

#define DEV __device__ __forceinline__

constexpr int D_DIM = 1024;   // model dim
constexpr int F_DIM = 2816;   // ffn dim
constexpr int E_NUM = 8;      // experts
constexpr int A_TOK = 2048;   // B*S tokens

constexpr int TM   = 128;     // token tile (M)
constexpr int KC   = 64;      // K chunk staged in LDS
constexpr int TN_U = 64;      // N tile, up-proj (2 GEMMs -> narrower N)
constexpr int TN_D = 128;     // N tile, down-proj

typedef __attribute__((ext_vector_type(16))) __bf16 v16bf;
typedef __attribute__((ext_vector_type(8)))  float  v8f;
typedef unsigned int u32x4 __attribute__((ext_vector_type(4)));
typedef int          i32x4 __attribute__((ext_vector_type(4)));
typedef int          i32x8 __attribute__((ext_vector_type(8)));

union Frag {
  v16bf v;
  uint4 q[2];
};

// native fp32 -> bf16 (RNE) via hardware cvt (v_cvt_pk_bf16_f32)
DEV unsigned short f2bf(float f) {
  __bf16 h = (__bf16)f;
  return __builtin_bit_cast(unsigned short, h);
}
DEV unsigned int pack2bf(float a, float b) {
  return (unsigned int)f2bf(a) | ((unsigned int)f2bf(b) << 16);
}

// ---------------------------------------------------------------------------
// Tensor Data Mover probe (guarded: falls back to manual staging if absent)
// ---------------------------------------------------------------------------
#if defined(__has_builtin)
#if __has_builtin(__builtin_amdgcn_tensor_load_to_lds) && \
    __has_builtin(__builtin_amdgcn_s_wait_tensorcnt)
#define HAVE_TDM 1
#endif
#endif
#ifndef HAVE_TDM
#define HAVE_TDM 0
#endif
constexpr bool kHaveTDM = (HAVE_TDM != 0);

typedef __attribute__((address_space(3))) void lds_void_t;
DEV unsigned lds_offset_of(void* p) {
  return (unsigned)(unsigned long long)(lds_void_t*)p;
}

#if HAVE_TDM
// 2D tile DMA: global (bf16, row pitch = pitch_elems) -> LDS, rows x cols elems
DEV void tdm_load_2d_bf16(const void* gptr, unsigned lds_byte_addr,
                          int cols, int rows, int pitch_elems,
                          int tensor_cols, int tensor_rows) {
  const unsigned long long ga = (unsigned long long)gptr;
  u32x4 g0;
  g0[0] = 1u;                                             // count=1, no gather
  g0[1] = lds_byte_addr;                                  // lds_addr (bytes)
  g0[2] = (unsigned)(ga & 0xFFFFFFFFu);                   // global_addr[31:0]
  g0[3] = (unsigned)((ga >> 32) & 0x01FFFFFFu) | (2u << 30);  // addr[56:32]|type=2
  i32x8 g1 = {};
  g1[0] = (1 << 16);                                      // data_size = 2 bytes
  g1[1] = (int)(((unsigned)tensor_cols & 0xFFFFu) << 16); // tensor_dim0[15:0]
  g1[2] = (int)((((unsigned)tensor_cols >> 16) & 0xFFFFu) |
                (((unsigned)tensor_rows & 0xFFFFu) << 16)); // dim0 hi | dim1 lo
  g1[3] = (int)((((unsigned)tensor_rows >> 16) & 0xFFFFu) |
                (((unsigned)cols & 0xFFFFu) << 16));      // dim1 hi | tile_dim0
  g1[4] = rows & 0xFFFF;                                  // tile_dim1 (tile_dim2=0)
  g1[5] = pitch_elems;                                    // tensor_dim0_stride lo
  g1[6] = 0;                                              // stride hi | dim1_stride
  g1[7] = 0;
  i32x4 gz = {};
#if __has_include(<hip/amd_detail/amd_gfx1250_TDM.h>)
  i32x8 gz8 = {};
  __builtin_amdgcn_tensor_load_to_lds(g0, g1, gz, gz, gz8, 0);   // 6-arg toolchain
#else
  __builtin_amdgcn_tensor_load_to_lds(g0, g1, gz, gz, 0);        // 5-arg ROCm 7.2
#endif
}
DEV void tdm_wait() { __builtin_amdgcn_s_wait_tensorcnt(0); }
#else
DEV void tdm_load_2d_bf16(const void*, unsigned, int, int, int, int, int) {}
DEV void tdm_wait() {}
#endif

// ---------------------------------------------------------------------------
// Kernel 1: router -- one wave32 per token, top-1 expert + sigmoid gate
// ---------------------------------------------------------------------------
__global__ __launch_bounds__(256)
void router_kernel(const float* __restrict__ x, const float* __restrict__ router,
                   int* __restrict__ expertId, float* __restrict__ gates) {
  const int wave = threadIdx.x >> 5;
  const int lane = threadIdx.x & 31;
  const int t = blockIdx.x * 8 + wave;
  if (t >= A_TOK) return;

  float acc[E_NUM];
#pragma unroll
  for (int e = 0; e < E_NUM; ++e) acc[e] = 0.0f;

  const float* xp = x + (size_t)t * D_DIM;
  for (int d = lane; d < D_DIM; d += 32) {
    const float xv = xp[d];
    const float* rp = router + (size_t)d * E_NUM;
#pragma unroll
    for (int e = 0; e < E_NUM; ++e) acc[e] += xv * rp[e];
  }
#pragma unroll
  for (int e = 0; e < E_NUM; ++e) {
#pragma unroll
    for (int off = 16; off > 0; off >>= 1)
      acc[e] += __shfl_xor(acc[e], off, 32);
  }
  if (lane == 0) {
    int best = 0;
    float bv = acc[0];
#pragma unroll
    for (int e = 1; e < E_NUM; ++e)
      if (acc[e] > bv) { bv = acc[e]; best = e; }
    expertId[t] = best;
    gates[t] = 1.0f / (1.0f + __expf(-bv));
  }
}

// ---------------------------------------------------------------------------
// Kernel 2: group tokens into per-expert lists
// ---------------------------------------------------------------------------
__global__ __launch_bounds__(256)
void group_kernel(const int* __restrict__ expertId, int* __restrict__ counts,
                  int* __restrict__ list) {
  const int t = blockIdx.x * blockDim.x + threadIdx.x;
  if (t < A_TOK) {
    const int e = expertId[t];
    const int pos = atomicAdd(&counts[e], 1);
    list[e * A_TOK + pos] = t;
  }
}

// ---------------------------------------------------------------------------
// Kernel 3: up-projection + SwiGLU: mid[t,:] = silu(g*x @ W1) * (g*x @ W3)
// 256 thr = 8 waves; tile 128(M) x 64(N); K ping-pong staged 64-wide in LDS.
// ---------------------------------------------------------------------------
template <bool ROUTED>
__global__ __launch_bounds__(256)
void gemm_swiglu(const float* __restrict__ x,
                 const float* __restrict__ W1, const float* __restrict__ W3,
                 const int* __restrict__ list, const int* __restrict__ counts,
                 const float* __restrict__ gates,
                 unsigned short* __restrict__ mid) {
  __shared__ __align__(16) unsigned short Abuf[2][TM * KC];     // 16 KB x2
  __shared__ __align__(16) unsigned short B1buf[2][TN_U * KC];  //  8 KB x2
  __shared__ __align__(16) unsigned short B3buf[2][TN_U * KC];  //  8 KB x2
  __shared__ int   tokLDS[TM];
  __shared__ float gateLDS[TM];

  const int tid = threadIdx.x;
  const int nbase = blockIdx.x * TN_U;

  int e, mtile;
  if constexpr (ROUTED) { e = (int)blockIdx.y >> 4; mtile = (int)blockIdx.y & 15; }
  else                  { e = 0;                    mtile = (int)blockIdx.y; }

  const int cnt = ROUTED ? counts[e] : A_TOK;
  if (mtile * TM >= cnt) return;            // uniform early exit (no barriers yet)
  const int mbase = mtile * TM;

  const size_t wofs = ROUTED ? (size_t)e * D_DIM * F_DIM : 0;
  const float* w1p = W1 + wofs;
  const float* w3p = W3 + wofs;

  if (tid < TM) {
    int token = -1;
    float g = 0.0f;
    if (mbase + tid < cnt) {
      if constexpr (ROUTED) { token = list[e * A_TOK + mbase + tid]; g = gates[token]; }
      else                  { token = mbase + tid;                   g = 1.0f; }
    }
    tokLDS[tid] = token;
    gateLDS[tid] = g;
  }
  __syncthreads();

  const int wave = tid >> 5, lane = tid & 31;
  const int hi = lane >> 4, lo = lane & 15;

  v8f acc1[4] = {}, acc3[4] = {};

  auto stage = [&](int sel, int ks) {
    unsigned short* A  = Abuf[sel];
    unsigned short* B1 = B1buf[sel];
    unsigned short* B3 = B3buf[sel];
    // A: 128 rows x 64 K fp32 -> gate-scaled bf16 (2048 float4, 8/thread)
#pragma unroll
    for (int i = 0; i < 8; ++i) {
      const int flat = tid + 256 * i;
      const int row = flat >> 4;
      const int c4  = (flat & 15) * 4;
      const int token = tokLDS[row];
      float4 v = make_float4(0.f, 0.f, 0.f, 0.f);
      if (token >= 0)
        v = *(const float4*)(x + (size_t)token * D_DIM + ks + c4);
      const float g = gateLDS[row];
      *(uint2*)&A[row * KC + c4] =
          make_uint2(pack2bf(v.x * g, v.y * g), pack2bf(v.z * g, v.w * g));
    }
    // B1/B3: 64 K-rows x 64 N fp32 -> bf16, transposed [n][k] (1024 float4, 4/thr)
#pragma unroll
    for (int i = 0; i < 4; ++i) {
      const int flat = tid + 256 * i;
      const int kr = flat >> 4;
      const int n4 = (flat & 15) * 4;
      const size_t gofs = (size_t)(ks + kr) * F_DIM + nbase + n4;
      const float4 b1 = *(const float4*)(w1p + gofs);
      const float4 b3 = *(const float4*)(w3p + gofs);
      if (i == 0 && ks + 2 * KC < D_DIM) {   // prefetch stage s+2
        __builtin_prefetch(w1p + gofs + (size_t)(2 * KC) * F_DIM, 0, 0);
        __builtin_prefetch(w3p + gofs + (size_t)(2 * KC) * F_DIM, 0, 0);
      }
      B1[(n4 + 0) * KC + kr] = f2bf(b1.x);
      B1[(n4 + 1) * KC + kr] = f2bf(b1.y);
      B1[(n4 + 2) * KC + kr] = f2bf(b1.z);
      B1[(n4 + 3) * KC + kr] = f2bf(b1.w);
      B3[(n4 + 0) * KC + kr] = f2bf(b3.x);
      B3[(n4 + 1) * KC + kr] = f2bf(b3.y);
      B3[(n4 + 2) * KC + kr] = f2bf(b3.z);
      B3[(n4 + 3) * KC + kr] = f2bf(b3.w);
    }
  };

  auto compute = [&](int sel) {
    const unsigned short* A  = Abuf[sel];
    const unsigned short* B1 = B1buf[sel];
    const unsigned short* B3 = B3buf[sel];
#pragma unroll
    for (int kk = 0; kk < KC; kk += 32) {
      Frag a;
      const int row = 16 * wave + lo;
      a.q[0] = *(const uint4*)&A[row * KC + kk + 16 * hi];
      a.q[1] = *(const uint4*)&A[row * KC + kk + 16 * hi + 8];
#pragma unroll
      for (int j = 0; j < 4; ++j) {
        const int n = 16 * j + lo;
        Frag b1, b3;
        b1.q[0] = *(const uint4*)&B1[n * KC + kk + 16 * hi];
        b1.q[1] = *(const uint4*)&B1[n * KC + kk + 16 * hi + 8];
        b3.q[0] = *(const uint4*)&B3[n * KC + kk + 16 * hi];
        b3.q[1] = *(const uint4*)&B3[n * KC + kk + 16 * hi + 8];
        acc1[j] = __builtin_amdgcn_wmma_f32_16x16x32_bf16(
            false, a.v, false, b1.v, (short)0, acc1[j], false, false);
        acc3[j] = __builtin_amdgcn_wmma_f32_16x16x32_bf16(
            false, a.v, false, b3.v, (short)0, acc3[j], false, false);
      }
    }
  };

  constexpr int NS = D_DIM / KC;   // 16 stages
  stage(0, 0);
  __syncthreads();
  for (int s = 0; s < NS; ++s) {
    if (s + 1 < NS) stage((s + 1) & 1, (s + 1) * KC);   // global->LDS for s+1
    compute(s & 1);                                     // WMMA on stage s
    __syncthreads();
  }

  // epilogue: SwiGLU, write bf16 mid[token, col]
#pragma unroll
  for (int j = 0; j < 4; ++j) {
    const int col = nbase + 16 * j + lo;
#pragma unroll
    for (int r = 0; r < 8; ++r) {
      const int m = 16 * wave + r + 8 * hi;
      const int token = tokLDS[m];
      if (token < 0) continue;
      const float h1 = acc1[j][r];
      const float val = (h1 / (1.0f + __expf(-h1))) * acc3[j][r];
      mid[(size_t)token * F_DIM + col] = f2bf(val);
    }
  }
}

// ---------------------------------------------------------------------------
// Kernel 4: down-projection: out[t,:] (+)= mid[t,:] @ W2   (K = F_DIM)
// tile 128(M) x 128(N). Shared-expert path stages the contiguous bf16 A tile
// with the Tensor Data Mover (one wave issues DMA; TENSORcnt-waited).
// ---------------------------------------------------------------------------
template <bool ROUTED>
__global__ __launch_bounds__(256)
void gemm_down(const unsigned short* __restrict__ mid,
               const float* __restrict__ W2,
               const int* __restrict__ list, const int* __restrict__ counts,
               float* __restrict__ out) {
  __shared__ __align__(16) unsigned short Abuf[2][TM * KC];    // 16 KB x2
  __shared__ __align__(16) unsigned short Bbuf[2][TN_D * KC];  // 16 KB x2
  __shared__ int tokLDS[TM];

  constexpr bool useTDM = (!ROUTED) && kHaveTDM;

  const int tid = threadIdx.x;
  const int nbase = blockIdx.x * TN_D;

  int e, mtile;
  if constexpr (ROUTED) { e = (int)blockIdx.y >> 4; mtile = (int)blockIdx.y & 15; }
  else                  { e = 0;                    mtile = (int)blockIdx.y; }

  const int cnt = ROUTED ? counts[e] : A_TOK;
  if (mtile * TM >= cnt) return;
  const int mbase = mtile * TM;

  const float* w2p = W2 + (ROUTED ? (size_t)e * F_DIM * D_DIM : 0);

  if (tid < TM) {
    int token = -1;
    if (mbase + tid < cnt) {
      if constexpr (ROUTED) token = list[e * A_TOK + mbase + tid];
      else                  token = mbase + tid;
    }
    tokLDS[tid] = token;
  }
  __syncthreads();

  const int wave = tid >> 5, lane = tid & 31;
  const int hi = lane >> 4, lo = lane & 15;

  v8f acc[8] = {};

  auto stage = [&](int sel, int ks) {
    unsigned short* A = Abuf[sel];
    unsigned short* B = Bbuf[sel];
    // A tile: TM rows x KC bf16
    if constexpr (useTDM) {
      // contiguous rows (shared expert): single TDM 2D tile, wave 0 issues
      if (wave == 0)
        tdm_load_2d_bf16(mid + (size_t)mbase * F_DIM + ks, lds_offset_of(A),
                         /*cols=*/KC, /*rows=*/TM, /*pitch=*/F_DIM,
                         /*tensor_cols=*/F_DIM, /*tensor_rows=*/A_TOK);
    } else {
      // gathered rows via token list (1024 uint4, 4/thread)
#pragma unroll
      for (int i = 0; i < 4; ++i) {
        const int flat = tid + 256 * i;
        const int row = flat >> 3;
        const int c8  = (flat & 7) * 8;
        const int token = tokLDS[row];
        uint4 v = make_uint4(0u, 0u, 0u, 0u);
        if (token >= 0)
          v = *(const uint4*)(mid + (size_t)token * F_DIM + ks + c8);
        *(uint4*)&A[row * KC + c8] = v;
      }
    }
    // B: 64 K-rows x 128 N fp32 -> bf16, transposed [n][k] (2048 float4, 8/thr)
#pragma unroll
    for (int i = 0; i < 8; ++i) {
      const int flat = tid + 256 * i;
      const int kr = flat >> 5;
      const int n4 = (flat & 31) * 4;
      const size_t gofs = (size_t)(ks + kr) * D_DIM + nbase + n4;
      const float4 b = *(const float4*)(w2p + gofs);
      if (i == 0 && ks + 2 * KC < F_DIM)
        __builtin_prefetch(w2p + gofs + (size_t)(2 * KC) * D_DIM, 0, 0);
      B[(n4 + 0) * KC + kr] = f2bf(b.x);
      B[(n4 + 1) * KC + kr] = f2bf(b.y);
      B[(n4 + 2) * KC + kr] = f2bf(b.z);
      B[(n4 + 3) * KC + kr] = f2bf(b.w);
    }
  };

  auto compute = [&](int sel) {
    const unsigned short* A = Abuf[sel];
    const unsigned short* B = Bbuf[sel];
#pragma unroll
    for (int kk = 0; kk < KC; kk += 32) {
      Frag a;
      const int row = 16 * wave + lo;
      a.q[0] = *(const uint4*)&A[row * KC + kk + 16 * hi];
      a.q[1] = *(const uint4*)&A[row * KC + kk + 16 * hi + 8];
#pragma unroll
      for (int j = 0; j < 8; ++j) {
        const int n = 16 * j + lo;
        Frag b;
        b.q[0] = *(const uint4*)&B[n * KC + kk + 16 * hi];
        b.q[1] = *(const uint4*)&B[n * KC + kk + 16 * hi + 8];
        acc[j] = __builtin_amdgcn_wmma_f32_16x16x32_bf16(
            false, a.v, false, b.v, (short)0, acc[j], false, false);
      }
    }
  };

  constexpr int NS = F_DIM / KC;   // 44 stages
  stage(0, 0);
  if constexpr (useTDM) { if (wave == 0) tdm_wait(); }
  __syncthreads();
  for (int s = 0; s < NS; ++s) {
    if (s + 1 < NS) stage((s + 1) & 1, (s + 1) * KC);
    compute(s & 1);
    if constexpr (useTDM) { if (wave == 0) tdm_wait(); }  // DMA done before barrier
    __syncthreads();
  }

#pragma unroll
  for (int j = 0; j < 8; ++j) {
    const int col = nbase + 16 * j + lo;
#pragma unroll
    for (int r = 0; r < 8; ++r) {
      const int m = 16 * wave + r + 8 * hi;
      const int token = tokLDS[m];
      if (token < 0) continue;
      float* p = out + (size_t)token * D_DIM + col;
      if constexpr (ROUTED) *p += acc[j][r];
      else                  *p  = acc[j][r];
    }
  }
}

// ---------------------------------------------------------------------------
extern "C" void kernel_launch(void* const* d_in, const int* in_sizes, int n_in,
                              void* d_out, int out_size, void* d_ws, size_t ws_size,
                              hipStream_t stream) {
  (void)in_sizes; (void)n_in; (void)out_size; (void)ws_size;

  const float* x      = (const float*)d_in[0];  // [a, D]
  const float* router = (const float*)d_in[1];  // [D, E]
  const float* w1     = (const float*)d_in[2];  // [E, D, F]
  const float* w3     = (const float*)d_in[3];  // [E, D, F]
  const float* w2     = (const float*)d_in[4];  // [E, F, D]
  const float* ws1    = (const float*)d_in[5];  // [D, F]
  const float* ws3    = (const float*)d_in[6];  // [D, F]
  const float* ws2    = (const float*)d_in[7];  // [F, D]
  // d_in[8] = top_k (==1, handled statically)
  float* out = (float*)d_out;

  char* ws = (char*)d_ws;
  int*            counts   = (int*)ws;                             // 32 B
  int*            expertId = (int*)(ws + 256);                     // 8 KB
  float*          gates    = (float*)(ws + 256 + 8192);            // 8 KB
  int*            list     = (int*)(ws + 256 + 16384);             // 64 KB
  unsigned short* mid      = (unsigned short*)(ws + 82176);        // a*F bf16, 11.5 MB

  hipMemsetAsync(counts, 0, E_NUM * sizeof(int), stream);
  router_kernel<<<A_TOK / 8, 256, 0, stream>>>(x, router, expertId, gates);
  group_kernel<<<A_TOK / 256, 256, 0, stream>>>(expertId, counts, list);

  // Shared expert: write path
  gemm_swiglu<false><<<dim3(F_DIM / TN_U, A_TOK / TM), 256, 0, stream>>>(
      x, ws1, ws3, nullptr, nullptr, nullptr, mid);
  gemm_down<false><<<dim3(D_DIM / TN_D, A_TOK / TM), 256, 0, stream>>>(
      mid, ws2, nullptr, nullptr, out);

  // Routed experts (top-1): accumulate path; max grid, blocks early-exit
  gemm_swiglu<true><<<dim3(F_DIM / TN_U, E_NUM * (A_TOK / TM)), 256, 0, stream>>>(
      x, w1, w3, list, counts, gates, mid);
  gemm_down<true><<<dim3(D_DIM / TN_D, E_NUM * (A_TOK / TM)), 256, 0, stream>>>(
      mid, w2, list, counts, out);
}
